// precond_net_82798379532799
// MI455X (gfx1250) — matrix operand order repaired and checked by reading, not compiled, not run on hardware
//
#include <hip/hip_runtime.h>

typedef __attribute__((ext_vector_type(16))) _Float16 v16h;
typedef __attribute__((ext_vector_type(8)))  _Float16 v8h;
typedef __attribute__((ext_vector_type(8)))  float    v8f;

#define HP  514           // halo-padded height/width
#define IMG 512

// ---------------------------------------------------------------------------
// helpers
// ---------------------------------------------------------------------------
__device__ __forceinline__ v8h prelu_cvt(v8f a, float p) {
  v8h r;
#pragma unroll
  for (int i = 0; i < 8; ++i) {
    float v = a[i];
    v = (v >= 0.f) ? v : p * v;
    r[i] = (_Float16)v;
  }
  return r;
}

// zero the 1-pixel halo border of a [HP][HP][C] f16 buffer (C multiple of 8)
__global__ void zero_borders(_Float16* __restrict__ buf, int C) {
  int chunksPerPix = C >> 3;
  int tid = blockIdx.x * blockDim.x + threadIdx.x;
  int total = (2 * HP + 2 * IMG) * chunksPerPix;   // 2052 border pixels
  if (tid >= total) return;
  int bp = tid / chunksPerPix;
  int ch = (tid % chunksPerPix) << 3;
  int row, col;
  if (bp < HP)                { row = 0;                    col = bp; }
  else if (bp < 2 * HP)       { row = HP - 1;               col = bp - HP; }
  else if (bp < 2 * HP + IMG) { row = bp - 2 * HP + 1;      col = 0; }
  else                        { row = bp - 2 * HP - IMG + 1; col = HP - 1; }
  v8h z = {};
  *(v8h*)(buf + ((size_t)row * HP + col) * C + ch) = z;
}

// repack OIHW f32 weights into per-lane WMMA A-fragment order, f16.
// packed index: ((t*MT + m)*KT + kk)*512 + lane*16 + h
__global__ void pack_weights(const float* __restrict__ w, _Float16* __restrict__ wp,
                             int CIN, int COUT) {
  int KT = CIN >> 5, MT = COUT >> 4;
  int n = 9 * MT * KT * 512;
  int idx = blockIdx.x * blockDim.x + threadIdx.x;
  if (idx >= n) return;
  int h    = idx & 15;
  int lane = (idx >> 4) & 31;
  int blk  = idx >> 9;
  int kk = blk % KT;
  int m  = (blk / KT) % MT;
  int t  = blk / (KT * MT);
  int M = m * 16 + (lane & 15);
  int K = kk * 32 + ((lane >> 4) << 4) + h;
  int ky = t / 3, kx = t % 3;
  float v = w[(((size_t)M * CIN + K) * 3 + ky) * 3 + kx];
  wp[idx] = (_Float16)v;
}

// ---------------------------------------------------------------------------
// layer 0: 1 -> 64, 1x1 conv + PReLU, f32 in -> f16 NHWC halo out
// ---------------------------------------------------------------------------
__global__ void layer0_k(const float* __restrict__ A, const float* __restrict__ w,
                         const float* __restrict__ b, const float* __restrict__ p,
                         _Float16* __restrict__ out) {
  int idx = blockIdx.x * blockDim.x + threadIdx.x;
  if (idx >= IMG * IMG) return;
  int y = idx >> 9, x = idx & 511;
  float v = A[idx];
  float pp = p[0];
  _Float16* op = out + ((size_t)(y + 1) * HP + (x + 1)) * 64;
#pragma unroll 8
  for (int c = 0; c < 64; ++c) {
    float r = w[c] * v + b[c];
    r = (r >= 0.f) ? r : pp * r;
    op[c] = (_Float16)r;
  }
}

// ---------------------------------------------------------------------------
// middle layers: 3x3 conv as 9 shifted GEMMs via WMMA f32 = f16 x f16.
// One wave owns a 2x2 tile of (cout-tile x pixel-tile): 32 cout x 32 pixels.
// The 3x34xCin input patch is staged into LDS in 64-channel chunks with
// async global->LDS copies (ASYNCcnt), double buffered, WMMA hides latency.
// ---------------------------------------------------------------------------
template <int CIN, int COUT>
__global__ __launch_bounds__(256)
void conv3_wmma(const _Float16* __restrict__ in, const _Float16* __restrict__ wpk,
                const float* __restrict__ bias, const float* __restrict__ prelu,
                _Float16* __restrict__ out) {
  constexpr int KT = CIN >> 5;    // k-steps of 32
  constexpr int MT = COUT >> 4;   // cout tiles of 16
  constexpr int NC = CIN >> 6;    // 64-channel staging chunks

  // double-buffered patch: [3 rows][34 cols][64 ch] f16 = 13 KB per buffer
  __shared__ __attribute__((aligned(128))) _Float16 lbuf[2][3 * 34 * 64];

  const int tid  = threadIdx.x;
  const int lane = tid & 31;
  const int wave = tid >> 5;
  const int wavesPerBlock = blockDim.x >> 5;
  const int nthreads = blockDim.x;

  // pixel-pair tile: 32 contiguous pixels in one row
  const int pt = blockIdx.x;             // 0 .. 8191
  const int y  = pt >> 4;                // 0 .. 511
  const int x0 = (pt & 15) << 5;         // 0 .. 480 step 32
  const int kh = lane >> 4;              // which K-half of B this lane holds

  // cout tile pair
  const int mPair = blockIdx.y * wavesPerBlock + wave;
  const int m0 = mPair * 2;
  const int m1 = m0 + 1;

  v8f acc00, acc01, acc10, acc11;
  const int coBase0 = m0 * 16 + kh * 8;
  const int coBase1 = m1 * 16 + kh * 8;
#pragma unroll
  for (int r = 0; r < 8; ++r) {
    acc00[r] = bias[coBase0 + r];
    acc10[r] = bias[coBase1 + r];
  }
  acc01 = acc00;
  acc11 = acc10;

  // async stage of one 64-channel chunk of the 3x34 patch into lbuf[buf]
  // 816 x b128 transfers, spread across the block, tracked by ASYNCcnt
  auto stage = [&](int kc, int buf) {
    const _Float16* src = in + (size_t)kc * 64;
    for (int i = tid; i < 816; i += nthreads) {
      int ch = (i & 7) << 3;                // half offset within 64 channels
      int pc = i >> 3;                      // 0..101 = row*34+col of patch
      int r  = pc / 34;
      int c  = pc - r * 34;
      const _Float16* gp = src + ((size_t)(y + r) * HP + (x0 + c)) * CIN + ch;
      unsigned loff = (unsigned)(uintptr_t)(&lbuf[buf][pc * 64 + ch]);
      asm volatile("global_load_async_to_lds_b128 %0, %1, off"
                   :: "v"(loff), "v"((unsigned long long)(uintptr_t)gp)
                   : "memory");
    }
  };

  stage(0, 0);
  asm volatile("s_wait_asynccnt 0x0" ::: "memory");
  __syncthreads();

  int buf = 0;
#pragma unroll 1
  for (int kc = 0; kc < NC; ++kc) {
    if (kc + 1 < NC) stage(kc + 1, buf ^ 1);       // prefetch next chunk

    const _Float16* lb = &lbuf[buf][0];
#pragma unroll 1
    for (int t = 0; t < 9; ++t) {
      const int dy = t / 3 - 1, dx = t % 3 - 1;
      // B fragments from LDS: patch row dy+1, col (lane&15)+1+dx, K-half kh
      const _Float16* lp = lb + (((dy + 1) * 34 + (lane & 15) + 1 + dx) * 64) + kh * 16;
#pragma unroll
      for (int kk2 = 0; kk2 < 2; ++kk2) {
        const int kk = kc * 2 + kk2;
        v16h bA = *(const v16h*)(lp + kk2 * 32);            // pixels x0..x0+15
        v16h bB = *(const v16h*)(lp + 16 * 64 + kk2 * 32);  // pixels x0+16..x0+31
        v16h a0 = *(const v16h*)(wpk + (((size_t)t * MT + m0) * KT + kk) * 512 + lane * 16);
        v16h a1 = *(const v16h*)(wpk + (((size_t)t * MT + m1) * KT + kk) * 512 + lane * 16);
        acc00 = __builtin_amdgcn_wmma_f32_16x16x32_f16(false, a0, false, bA, (short)0, acc00, false, false);
        acc01 = __builtin_amdgcn_wmma_f32_16x16x32_f16(false, a0, false, bB, (short)0, acc01, false, false);
        acc10 = __builtin_amdgcn_wmma_f32_16x16x32_f16(false, a1, false, bA, (short)0, acc10, false, false);
        acc11 = __builtin_amdgcn_wmma_f32_16x16x32_f16(false, a1, false, bB, (short)0, acc11, false, false);
      }
    }

    asm volatile("s_wait_asynccnt 0x0" ::: "memory");  // own async copies landed
    __syncthreads();                                    // everyone's landed
    buf ^= 1;
  }

  const float p = prelu[0];
  const int pxA = x0 + (lane & 15);
  _Float16* opA = out + ((size_t)(y + 1) * HP + (pxA + 1)) * COUT;
  _Float16* opB = out + ((size_t)(y + 1) * HP + (pxA + 16 + 1)) * COUT;
  *(v8h*)(opA + coBase0) = prelu_cvt(acc00, p);
  *(v8h*)(opA + coBase1) = prelu_cvt(acc10, p);
  *(v8h*)(opB + coBase0) = prelu_cvt(acc01, p);
  *(v8h*)(opB + coBase1) = prelu_cvt(acc11, p);
}

// ---------------------------------------------------------------------------
// layer 5: 64 -> 1, 3x3 conv + PReLU  (f16 NHWC halo in -> f32 plane out)
// ---------------------------------------------------------------------------
__global__ void layer5_k(const _Float16* __restrict__ in, const float* __restrict__ w,
                         const float* __restrict__ b, const float* __restrict__ p,
                         float* __restrict__ ybuf) {
  int idx = blockIdx.x * blockDim.x + threadIdx.x;
  if (idx >= IMG * IMG) return;
  int y = idx >> 9, x = idx & 511;
  float s = b[0];
#pragma unroll
  for (int t = 0; t < 9; ++t) {
    int dy = t / 3 - 1, dx = t % 3 - 1;
    const _Float16* ip = in + ((size_t)(y + 1 + dy) * HP + (x + 1 + dx)) * 64;
#pragma unroll 8
    for (int c = 0; c < 64; ++c) s += (float)ip[c] * w[c * 9 + t];
  }
  float pp = p[0];
  s = (s >= 0.f) ? s : pp * s;
  ybuf[idx] = s;
}

// 0.5 * (y + y^T) over the spatial dims
__global__ void symmetrize_k(const float* __restrict__ ybuf, float* __restrict__ out) {
  int idx = blockIdx.x * blockDim.x + threadIdx.x;
  if (idx >= IMG * IMG) return;
  int i = idx >> 9, j = idx & 511;
  out[idx] = 0.5f * (ybuf[idx] + ybuf[(size_t)j * IMG + i]);
}

// ---------------------------------------------------------------------------
// launch
// ---------------------------------------------------------------------------
extern "C" void kernel_launch(void* const* d_in, const int* in_sizes, int n_in,
                              void* d_out, int out_size, void* d_ws, size_t ws_size,
                              hipStream_t stream) {
  (void)in_sizes; (void)n_in; (void)out_size; (void)ws_size;
  const float* A = (const float*)d_in[0];
  const float *w[6], *b[6], *p[6];
  for (int i = 0; i < 6; ++i) {
    w[i] = (const float*)d_in[1 + 3 * i];
    b[i] = (const float*)d_in[2 + 3 * i];
    p[i] = (const float*)d_in[3 + 3 * i];
  }

  // workspace layout (256B aligned slabs)
  char* ws = (char*)d_ws;
  size_t off = 0;
  auto alloc = [&](size_t bytes) -> char* {
    char* r = ws + off;
    off += (bytes + 255) & ~(size_t)255;
    return r;
  };
  const size_t P = (size_t)HP * HP;
  _Float16* act0 = (_Float16*)alloc(P * 64  * 2);
  _Float16* act1 = (_Float16*)alloc(P * 256 * 2);
  _Float16* act2 = (_Float16*)alloc(P * 512 * 2);
  _Float16* act3 = (_Float16*)alloc(P * 256 * 2);
  _Float16* act4 = (_Float16*)alloc(P * 64  * 2);
  _Float16* wpk1 = (_Float16*)alloc((size_t)9 * 16 * 2  * 512 * 2);  // 64->256
  _Float16* wpk2 = (_Float16*)alloc((size_t)9 * 32 * 8  * 512 * 2);  // 256->512
  _Float16* wpk3 = (_Float16*)alloc((size_t)9 * 16 * 16 * 512 * 2);  // 512->256
  _Float16* wpk4 = (_Float16*)alloc((size_t)9 * 4  * 8  * 512 * 2);  // 256->64
  float*    ybuf = (float*)   alloc((size_t)IMG * IMG * 4);

  const int T = 256;
  auto cdiv = [](int a, int bb) { return (a + bb - 1) / bb; };

  // 1) zero halo borders (interiors are fully rewritten each call)
  { int n = 2052 * (64  >> 3); zero_borders<<<cdiv(n, T), T, 0, stream>>>(act0, 64);  }
  { int n = 2052 * (256 >> 3); zero_borders<<<cdiv(n, T), T, 0, stream>>>(act1, 256); }
  { int n = 2052 * (512 >> 3); zero_borders<<<cdiv(n, T), T, 0, stream>>>(act2, 512); }
  { int n = 2052 * (256 >> 3); zero_borders<<<cdiv(n, T), T, 0, stream>>>(act3, 256); }
  { int n = 2052 * (64  >> 3); zero_borders<<<cdiv(n, T), T, 0, stream>>>(act4, 64);  }

  // 2) repack weights for WMMA A-fragments
  { int n = 9 * 16 * 2  * 512; pack_weights<<<cdiv(n, T), T, 0, stream>>>(w[1], wpk1, 64,  256); }
  { int n = 9 * 32 * 8  * 512; pack_weights<<<cdiv(n, T), T, 0, stream>>>(w[2], wpk2, 256, 512); }
  { int n = 9 * 16 * 16 * 512; pack_weights<<<cdiv(n, T), T, 0, stream>>>(w[3], wpk3, 512, 256); }
  { int n = 9 * 4  * 8  * 512; pack_weights<<<cdiv(n, T), T, 0, stream>>>(w[4], wpk4, 256, 64);  }

  // 3) layer chain
  layer0_k<<<cdiv(IMG * IMG, T), T, 0, stream>>>(A, w[0], b[0], p[0], act0);

  {
    dim3 g(8192, 1);  // MT=16 -> 8 mPairs -> 8 waves
    conv3_wmma<64, 256><<<g, 256, 0, stream>>>(act0, wpk1, b[1], p[1], act1);
  }
  {
    dim3 g(8192, 2);  // MT=32 -> 16 mPairs -> 8 waves x 2
    conv3_wmma<256, 512><<<g, 256, 0, stream>>>(act1, wpk2, b[2], p[2], act2);
  }
  {
    dim3 g(8192, 1);
    conv3_wmma<512, 256><<<g, 256, 0, stream>>>(act2, wpk3, b[3], p[3], act3);
  }
  {
    dim3 g(8192, 1);  // MT=4 -> 2 mPairs -> 2 waves
    conv3_wmma<256, 64><<<g, 64, 0, stream>>>(act3, wpk4, b[4], p[4], act4);
  }

  layer5_k<<<cdiv(IMG * IMG, T), T, 0, stream>>>(act4, w[5], b[5], p[5], ybuf);
  symmetrize_k<<<cdiv(IMG * IMG, T), T, 0, stream>>>(ybuf, (float*)d_out);
}
